// MultiHeadSelfAttention_22333829940058
// MI455X (gfx1250) — matrix operand order repaired and checked by reading, not compiled, not run on hardware
//
#include <hip/hip_runtime.h>

typedef __attribute__((ext_vector_type(16))) _Float16 v16h;
typedef __attribute__((ext_vector_type(8)))  float    v8f;

#define BB 4
#define TT 2048
#define DD 1024
#define HH 16
#define KH 64
#define HK 1024
#define BT (BB*TT)

static __device__ __forceinline__ v8f wmma(const v16h a, const v16h b, const v8f c) {
  return __builtin_amdgcn_wmma_f32_16x16x32_f16(false, a, false, b, (short)0, c, false, false);
}

// A-fragment (16x32 f16): lane m = lane%16, half-group hg = lane/16.
// element e -> k = hg*16 + e: one contiguous 16-half run (two 16B loads).
static __device__ __forceinline__ v16h load_a_frag(const _Float16* tile, int ld) {
  const int lane = threadIdx.x & 31;
  const int m  = lane & 15;
  const int hg = lane >> 4;
  const _Float16* p = tile + (size_t)m * ld + hg * 16;
  union { v16h v; uint4 q[2]; } u;
  u.q[0] = *reinterpret_cast<const uint4*>(p);
  u.q[1] = *reinterpret_cast<const uint4*>(p + 8);
  return u.v;
}

// B-fragment where the lane's 16 halves are contiguous in memory (32B load).
static __device__ __forceinline__ v16h load_b16(const _Float16* p) {
  return *reinterpret_cast<const v16h*>(p);
}

// ---------------- stage 0: casts / weight packing ----------------

__global__ void cast_f32_f16(const float* __restrict__ src, _Float16* __restrict__ dst, int n) {
  int i = blockIdx.x * blockDim.x + threadIdx.x;
  const int stride = gridDim.x * blockDim.x;
  for (; i < n; i += stride) dst[i] = (_Float16)src[i];
}

// Pack W[h][d][n] (H,D,K) into fragment-major f16:
// linear idx = ((h*4 + nt)*32 + dc)*512 + lane*16 + e
// value = W[h][ d = dc*32 + (lane/16)*16 + e ][ n = nt*16 + lane%16 ]
__global__ void pack_w_qkv(const float* __restrict__ W, _Float16* __restrict__ out) {
  const int idx  = blockIdx.x * blockDim.x + threadIdx.x;   // 0 .. 1048575
  const int e    = idx & 15;
  const int lane = (idx >> 4) & 31;
  const int dc   = (idx >> 9) & 31;
  const int nt   = (idx >> 14) & 3;
  const int h    = idx >> 16;
  const int n = nt * 16 + (lane & 15);
  const int d = dc * 32 + (lane >> 4) * 16 + e;
  out[idx] = (_Float16)W[((size_t)h * DD + d) * KH + n];
}

// Pack Wo (out = x @ Wo^T): B[i][j] = Wo[j][i].
// linear idx = (jt*32 + ic)*512 + lane*16 + e
__global__ void pack_wo(const float* __restrict__ Wo, _Float16* __restrict__ out) {
  const int idx  = blockIdx.x * blockDim.x + threadIdx.x;   // 0 .. 1048575
  const int e    = idx & 15;
  const int lane = (idx >> 4) & 31;
  const int ic   = (idx >> 9) & 31;
  const int jt   = idx >> 14;
  const int j = jt * 16 + (lane & 15);
  const int i = ic * 32 + (lane >> 4) * 16 + e;
  out[idx] = (_Float16)Wo[(size_t)j * HK + i];
}

// ---------------- stage 1: QKV projections ----------------
// grid = (BT/256, 3*H), block 256 (8 waves; each wave computes 32 rows x 64 cols).
// Q,K written [b,h][T,64] row-major; V written transposed [b,h][64,T].
__global__ __launch_bounds__(256) void qkv_gemm(
    const _Float16* __restrict__ Xh, const _Float16* __restrict__ Wp,
    _Float16* __restrict__ Qo, _Float16* __restrict__ Ko, _Float16* __restrict__ Vt)
{
  const int p = blockIdx.y / HH, h = blockIdx.y % HH;
  const int wave = threadIdx.x >> 5, lane = threadIdx.x & 31;
  const int m16 = lane & 15, hg = lane >> 4;
  const int row0 = blockIdx.x * 256 + wave * 32;
  const _Float16* Ab = Xh + (size_t)row0 * DD;
  const _Float16* Bb = Wp + (size_t)p * (HH * DD * KH) + (size_t)(h * 4) * 32 * 512;
  const v8f vzero = {};
  v8f c[2][4];
#pragma unroll
  for (int r = 0; r < 2; ++r)
#pragma unroll
    for (int n = 0; n < 4; ++n) c[r][n] = vzero;

  for (int dc = 0; dc < DD / 32; ++dc) {
    const v16h a0 = load_a_frag(Ab + dc * 32, DD);
    const v16h a1 = load_a_frag(Ab + (size_t)16 * DD + dc * 32, DD);
#pragma unroll
    for (int n = 0; n < 4; ++n) {
      const v16h bf = load_b16(Bb + (size_t)(n * 32 + dc) * 512 + lane * 16);
      c[0][n] = wmma(a0, bf, c[0][n]);
      c[1][n] = wmma(a1, bf, c[1][n]);
    }
  }
  const int b = row0 / TT;
  const int t0 = row0 % TT;
  const int bh = b * HH + h;
  if (p < 2) {
    _Float16* dst = (p == 0 ? Qo : Ko) + (size_t)bh * TT * KH;
#pragma unroll
    for (int r = 0; r < 2; ++r)
#pragma unroll
      for (int n = 0; n < 4; ++n)
#pragma unroll
        for (int v = 0; v < 8; ++v) {
          const int t = t0 + r * 16 + hg * 8 + v;
          dst[(size_t)t * KH + n * 16 + m16] = (_Float16)c[r][n][v];
        }
  } else {
    _Float16* dst = Vt + (size_t)bh * KH * TT;
#pragma unroll
    for (int r = 0; r < 2; ++r)
#pragma unroll
      for (int n = 0; n < 4; ++n)
#pragma unroll
        for (int v = 0; v < 8; ++v) {
          const int t = t0 + r * 16 + hg * 8 + v;
          dst[(size_t)(n * 16 + m16) * TT + t] = (_Float16)c[r][n][v];
        }
  }
}

// ---------------- stage 2: causal flash attention ----------------
// grid = (T/128, B*H), block 128 (4 waves; each wave owns 32 query rows).
// K/V fragments are loaded once per 32-key step and reused by both row-tiles.
__global__ __launch_bounds__(128) void attn_fwd(
    const _Float16* __restrict__ Q, const _Float16* __restrict__ Kb,
    const _Float16* __restrict__ Vt, _Float16* __restrict__ Oh)
{
  __shared__ __align__(32) _Float16 pbuf[4][2][16 * 32];
  const int bh = blockIdx.y;
  const int wave = threadIdx.x >> 5, lane = threadIdx.x & 31;
  const int m16 = lane & 15, hg = lane >> 4;
  const int t0 = blockIdx.x * 128 + wave * 32;
  const _Float16* Qb  = Q  + (size_t)bh * TT * KH;
  const _Float16* Kbb = Kb + (size_t)bh * TT * KH;
  const _Float16* Vb  = Vt + (size_t)bh * KH * TT;
  const v8f vzero = {};

  v16h qa[2][2];
#pragma unroll
  for (int r = 0; r < 2; ++r) {
    qa[r][0] = load_a_frag(Qb + (size_t)(t0 + 16 * r) * KH,      KH);
    qa[r][1] = load_a_frag(Qb + (size_t)(t0 + 16 * r) * KH + 32, KH);
  }
  v8f o[2][4];
  float mr[2][8], lr[2][8];
#pragma unroll
  for (int r = 0; r < 2; ++r) {
#pragma unroll
    for (int n = 0; n < 4; ++n) o[r][n] = vzero;
#pragma unroll
    for (int v = 0; v < 8; ++v) { mr[r][v] = -3.0e38f; lr[r][v] = 0.0f; }
  }

  const int send = t0 + 32;   // exclusive causal key bound for this wave's rows
  for (int sb = 0; sb < send; sb += 32) {
    // K fragments for the 2 key n-tiles x 2 head-dim chunks (shared by both row-tiles)
    v16h kf[2][2];
#pragma unroll
    for (int j = 0; j < 2; ++j)
#pragma unroll
      for (int cch = 0; cch < 2; ++cch)
        kf[j][cch] = load_b16(Kbb + (size_t)(sb + j * 16 + m16) * KH + cch * 32 + hg * 16);

#pragma unroll
    for (int r = 0; r < 2; ++r) {
      v8f s0 = wmma(qa[r][0], kf[0][0], vzero);
      s0     = wmma(qa[r][1], kf[0][1], s0);
      v8f s1 = wmma(qa[r][0], kf[1][0], vzero);
      s1     = wmma(qa[r][1], kf[1][1], s1);
#pragma unroll
      for (int v = 0; v < 8; ++v) {
        const int qrow = t0 + r * 16 + hg * 8 + v;
        float x0 = (sb + m16 > qrow)      ? -3.0e38f : s0[v] * 0.125f;
        float x1 = (sb + 16 + m16 > qrow) ? -3.0e38f : s1[v] * 0.125f;
        float rmax = fmaxf(x0, x1);
        rmax = fmaxf(rmax, __shfl_xor(rmax, 1));
        rmax = fmaxf(rmax, __shfl_xor(rmax, 2));
        rmax = fmaxf(rmax, __shfl_xor(rmax, 4));
        rmax = fmaxf(rmax, __shfl_xor(rmax, 8));
        const float mnew  = fmaxf(mr[r][v], rmax);
        const float alpha = __expf(mr[r][v] - mnew);
        mr[r][v] = mnew;
        const float p0 = __expf(x0 - mnew);
        const float p1 = __expf(x1 - mnew);
        float rs = p0 + p1;
        rs += __shfl_xor(rs, 1);
        rs += __shfl_xor(rs, 2);
        rs += __shfl_xor(rs, 4);
        rs += __shfl_xor(rs, 8);
        lr[r][v] = lr[r][v] * alpha + rs;
        o[r][0][v] *= alpha; o[r][1][v] *= alpha;
        o[r][2][v] *= alpha; o[r][3][v] *= alpha;
        pbuf[wave][r][(hg * 8 + v) * 32 + m16]      = (_Float16)p0;
        pbuf[wave][r][(hg * 8 + v) * 32 + 16 + m16] = (_Float16)p1;
      }
    }
    __asm__ volatile("s_wait_dscnt 0" ::: "memory");  // per-wave LDS RAW fence

    // V fragments for the 4 head-dim n-tiles (shared by both row-tiles)
    v16h vf[4];
#pragma unroll
    for (int n = 0; n < 4; ++n)
      vf[n] = load_b16(Vb + (size_t)(n * 16 + m16) * TT + sb + hg * 16);
#pragma unroll
    for (int r = 0; r < 2; ++r) {
      const v16h pa = load_a_frag(&pbuf[wave][r][0], 32);
#pragma unroll
      for (int n = 0; n < 4; ++n) o[r][n] = wmma(pa, vf[n], o[r][n]);
    }
  }

  const int b = bh / HH, h = bh % HH;
#pragma unroll
  for (int r = 0; r < 2; ++r)
#pragma unroll
    for (int v = 0; v < 8; ++v) {
      const int t = t0 + r * 16 + hg * 8 + v;
      const float inv = 1.0f / lr[r][v];
      _Float16* dst = Oh + (size_t)(b * TT + t) * HK + h * KH + m16;
      dst[0]  = (_Float16)(o[r][0][v] * inv);
      dst[16] = (_Float16)(o[r][1][v] * inv);
      dst[32] = (_Float16)(o[r][2][v] * inv);
      dst[48] = (_Float16)(o[r][3][v] * inv);
    }
}

// ---------------- stage 3: output projection (+bias, f32 out) ----------------
// grid = (BT/256, HK/64), block 256 (8 waves; each wave computes 32 rows x 64 cols).
__global__ __launch_bounds__(256) void out_proj(
    const _Float16* __restrict__ Oh, const _Float16* __restrict__ Wop,
    const float* __restrict__ bo, float* __restrict__ Y)
{
  const int wave = threadIdx.x >> 5, lane = threadIdx.x & 31;
  const int m16 = lane & 15, hg = lane >> 4;
  const int row0 = blockIdx.x * 256 + wave * 32;
  const int ntB = blockIdx.y * 4;
  const _Float16* Ab = Oh + (size_t)row0 * HK;
  const v8f vzero = {};
  v8f c[2][4];
#pragma unroll
  for (int r = 0; r < 2; ++r)
#pragma unroll
    for (int n = 0; n < 4; ++n) c[r][n] = vzero;

  for (int ic = 0; ic < HK / 32; ++ic) {
    const v16h a0 = load_a_frag(Ab + ic * 32, HK);
    const v16h a1 = load_a_frag(Ab + (size_t)16 * HK + ic * 32, HK);
#pragma unroll
    for (int n = 0; n < 4; ++n) {
      const v16h bf = load_b16(Wop + (size_t)((ntB + n) * 32 + ic) * 512 + lane * 16);
      c[0][n] = wmma(a0, bf, c[0][n]);
      c[1][n] = wmma(a1, bf, c[1][n]);
    }
  }
#pragma unroll
  for (int n = 0; n < 4; ++n) {
    const float bias = bo[(ntB + n) * 16 + m16];
#pragma unroll
    for (int r = 0; r < 2; ++r)
#pragma unroll
      for (int v = 0; v < 8; ++v) {
        const int row = row0 + r * 16 + hg * 8 + v;
        Y[(size_t)row * HK + (ntB + n) * 16 + m16] = c[r][n][v] + bias;
      }
  }
}

extern "C" void kernel_launch(void* const* d_in, const int* in_sizes, int n_in,
                              void* d_out, int out_size, void* d_ws, size_t ws_size,
                              hipStream_t stream)
{
  const float* X  = (const float*)d_in[0];
  const float* Wq = (const float*)d_in[1];
  const float* Wk = (const float*)d_in[2];
  const float* Wv = (const float*)d_in[3];
  const float* Wo = (const float*)d_in[4];
  const float* bo = (const float*)d_in[5];
  float* Y = (float*)d_out;

  _Float16* ws = (_Float16*)d_ws;
  const size_t M = 1024 * 1024;
  _Float16* Xh  = ws;                    // 8M halves
  _Float16* Qb  = ws + 8  * M;           // 8M
  _Float16* Kbf = ws + 16 * M;           // 8M
  _Float16* Vt  = ws + 24 * M;           // 8M (transposed V)
  _Float16* Oh  = ws + 32 * M;           // 8M
  _Float16* Wp  = ws + 40 * M;           // 3 x 1M packed Wq/Wk/Wv (contiguous)
  _Float16* WoP = ws + 43 * M;           // 1M packed Wo^T
  // total: 44M halves = 88 MB of d_ws

  cast_f32_f16<<<4096, 256, 0, stream>>>(X, Xh, BT * DD);
  pack_w_qkv<<<4096, 256, 0, stream>>>(Wq, Wp);
  pack_w_qkv<<<4096, 256, 0, stream>>>(Wk, Wp + 1 * M);
  pack_w_qkv<<<4096, 256, 0, stream>>>(Wv, Wp + 2 * M);
  pack_wo   <<<4096, 256, 0, stream>>>(Wo, WoP);
  qkv_gemm<<<dim3(BT / 256, 3 * HH), 256, 0, stream>>>(Xh, Wp, Qb, Kbf, Vt);
  attn_fwd<<<dim3(TT / 128, BB * HH), 128, 0, stream>>>(Qb, Kbf, Vt, Oh);
  out_proj<<<dim3(BT / 256, HK / 64), 256, 0, stream>>>(Oh, WoP, bo, Y);
}
